// GAT_ED_Transformer_89240830476324
// MI455X (gfx1250) — compile-verified, matrix operand order checked by reading, not run on hardware
//
#include <hip/hip_runtime.h>
#include <hip/hip_bf16.h>

// ---------------------------------------------------------------------------
// GAT + encoder/decoder transformer (seq_len==1 => MHA collapses to V,O GEMMs)
// Dense math via v_wmma_f32_16x16x32_f16. Operands stored fp16 in memory
// (converted once per buffer), fp32 accumulate, fp32 or fp16 epilogue.
// ---------------------------------------------------------------------------

typedef __attribute__((ext_vector_type(16))) _Float16 v16h;
typedef __attribute__((ext_vector_type(8)))  _Float16 v8h;
typedef __attribute__((ext_vector_type(8)))  float    v8f;

#define N_NODES 40000
#define N_EDGES 640000
#define NHEAD   4
#define NG      16
#define DMODEL  64
#define FFDIM   2048
#define CHUNK   4992   // FFN row chunk; 4992*8 + 64 = 40000, all %64==0

// ----------------------------- WMMA GEMM -----------------------------------
// C[M,N] = act(A[M,K] @ B[N,K]^T + bias). A,B fp16 in memory; fp32 acc.
// Requires M%64==0, N%64==0, K%64==0 (true for every call below).
// Fragment layout per CDNA5 ISA 7.12.2 (16-bit A 16x32): lanes 0-15 hold
// K={kh..kh+7, kh+16..kh+23} with kh=0, lanes 16-31 with kh=8. B mirrors A
// (per-lane column of B^T == row of W, K-contiguous).
__device__ inline v16h load_fragh(const _Float16* __restrict__ p) {
    v8h lo = *(const v8h*)(p);
    v8h hi = *(const v8h*)(p + 16);
    v16h r;
#pragma unroll
    for (int e = 0; e < 8; ++e) { r[e] = lo[e]; r[e + 8] = hi[e]; }
    return r;
}

template <typename OutT>
__global__ __launch_bounds__(128)
void wmma_gemm_nt(const _Float16* __restrict__ A, const _Float16* __restrict__ B,
                  OutT* __restrict__ C, const float* __restrict__ bias,
                  int M, int N, int K, int relu)
{
    const int wave = threadIdx.x >> 5;
    const int lane = threadIdx.x & 31;
    const int l16  = lane & 15;
    const int kh   = (lane >> 4) << 3;          // 0 or 8
    const int tM   = blockIdx.y * 64 + (wave >> 1) * 32;
    const int tN   = blockIdx.x * 64 + (wave & 1) * 32;

    v8f c00 = {}; v8f c01 = {}; v8f c10 = {}; v8f c11 = {};

    const _Float16* Ap0 = A + (size_t)(tM + l16) * K + kh;
    const _Float16* Ap1 = A + (size_t)(tM + 16 + l16) * K + kh;
    const _Float16* Bp0 = B + (size_t)(tN + l16) * K + kh;
    const _Float16* Bp1 = B + (size_t)(tN + 16 + l16) * K + kh;

    for (int kk = 0; kk < K; kk += 64) {        // K%64==0 -> 8 wmma per iter
#pragma unroll
        for (int u = 0; u < 2; ++u) {
            const int k0 = kk + u * 32;
            v16h a0 = load_fragh(Ap0 + k0);
            v16h a1 = load_fragh(Ap1 + k0);
            v16h b0 = load_fragh(Bp0 + k0);
            v16h b1 = load_fragh(Bp1 + k0);
            c00 = __builtin_amdgcn_wmma_f32_16x16x32_f16(false, a0, false, b0, (short)0, c00, false, false);
            c01 = __builtin_amdgcn_wmma_f32_16x16x32_f16(false, a0, false, b1, (short)0, c01, false, false);
            c10 = __builtin_amdgcn_wmma_f32_16x16x32_f16(false, a1, false, b0, (short)0, c10, false, false);
            c11 = __builtin_amdgcn_wmma_f32_16x16x32_f16(false, a1, false, b1, (short)0, c11, false, false);
        }
    }

    // C/D layout: lanes 0-15 -> N=lane, M=tile+r; lanes 16-31 -> N=lane-16, M=tile+8+r
    const int moff = (lane >> 4) << 3;
    auto stile = [&](v8f c, int tm, int tn) {
        const int n  = tn + l16;
        const float bv = bias ? bias[n] : 0.0f;
#pragma unroll
        for (int r = 0; r < 8; ++r) {
            float v = c[r] + bv;
            if (relu) v = fmaxf(v, 0.0f);
            C[(size_t)(tm + moff + r) * N + n] = (OutT)v;
        }
    };
    stile(c00, tM,      tN);
    stile(c01, tM,      tN + 16);
    stile(c10, tM + 16, tN);
    stile(c11, tM + 16, tN + 16);
}

// --------------------------- utility kernels -------------------------------
__global__ void fill_f32(float* __restrict__ p, float v, long long n) {
    long long i = (long long)blockIdx.x * blockDim.x + threadIdx.x;
    if (i < n) p[i] = v;
}

__global__ void cvt_f32_f16(_Float16* __restrict__ dst, const float* __restrict__ src,
                            long long n) {
    long long i = (long long)blockIdx.x * blockDim.x + threadIdx.x;
    if (i < n) dst[i] = (_Float16)src[i];
}

// src[R][C] fp32 -> dst[C][R] fp16
__global__ void transpose_cvt(_Float16* __restrict__ dst, const float* __restrict__ src,
                              int R, int Ccols) {
    int i = blockIdx.x * blockDim.x + threadIdx.x;
    if (i >= R * Ccols) return;
    int r = i / Ccols, c = i % Ccols;
    dst[(size_t)c * R + r] = (_Float16)src[i];
}

__global__ void bias_act(float* __restrict__ p, const float* __restrict__ b,
                         long long total, int cols, float slope) {
    long long i = (long long)blockIdx.x * blockDim.x + threadIdx.x;
    if (i >= total) return;
    float v = p[i] + b[(int)(i % cols)];
    p[i] = v >= 0.0f ? v : slope * v;
}

// ------------------------------- GAT ---------------------------------------
__device__ inline void atomicMaxF(float* addr, float v) {
    if (v >= 0.0f) atomicMax((int*)addr, __float_as_int(v));
    else           atomicMin((unsigned int*)addr, (unsigned int)__float_as_int(v));
}

__global__ void gat_scores(const float* __restrict__ h, const float* __restrict__ as,
                           const float* __restrict__ ad, float* __restrict__ ssrc,
                           float* __restrict__ sdst, int n, int oc) {
    int i = blockIdx.x * blockDim.x + threadIdx.x;
    if (i >= n * NHEAD) return;
    int node = i >> 2, hd = i & 3;
    const float* hp = h + (size_t)node * NHEAD * oc + hd * oc;
    const float* pa = as + hd * oc;
    const float* pb = ad + hd * oc;
    float s0 = 0.f, s1 = 0.f;
    for (int c = 0; c < oc; ++c) { float v = hp[c]; s0 += v * pa[c]; s1 += v * pb[c]; }
    ssrc[i] = s0; sdst[i] = s1;
}

__global__ void edge_max(const int* __restrict__ src, const int* __restrict__ dst,
                         const float* __restrict__ ssrc, const float* __restrict__ sdst,
                         float* __restrict__ m, int E, int Nn) {
    int i = blockIdx.x * blockDim.x + threadIdx.x;
    if (i >= (E + Nn) * NHEAD) return;
    int e = i >> 2, hd = i & 3, s, d;
    if (e < E) { s = src[e]; d = dst[e]; } else { s = d = e - E; }
    float v = ssrc[s * 4 + hd] + sdst[d * 4 + hd];
    v = v >= 0.0f ? v : 0.2f * v;
    atomicMaxF(m + d * 4 + hd, v);
}

__global__ void edge_exp(const int* __restrict__ src, const int* __restrict__ dst,
                         const float* __restrict__ ssrc, const float* __restrict__ sdst,
                         const float* __restrict__ m, float* __restrict__ ex,
                         float* __restrict__ den, int E, int Nn) {
    int i = blockIdx.x * blockDim.x + threadIdx.x;
    if (i >= (E + Nn) * NHEAD) return;
    int e = i >> 2, hd = i & 3, s, d;
    if (e < E) { s = src[e]; d = dst[e]; } else { s = d = e - E; }
    float v = ssrc[s * 4 + hd] + sdst[d * 4 + hd];
    v = v >= 0.0f ? v : 0.2f * v;
    float exv = expf(v - m[d * 4 + hd]);
    ex[i] = exv;
    atomicAdd(den + d * 4 + hd, exv);
}

__global__ void edge_aggr(const int* __restrict__ src, const int* __restrict__ dst,
                          const float* __restrict__ ex, const float* __restrict__ den,
                          const float* __restrict__ h, float* __restrict__ out,
                          int E, int Nn, int oc) {
    long long i = (long long)blockIdx.x * blockDim.x + threadIdx.x;
    const int C = NHEAD * oc;
    long long tot = (long long)(E + Nn) * C;
    if (i >= tot) return;
    int e  = (int)(i / C);
    int ch = (int)(i % C);
    int hd = ch / oc;
    int s, d;
    if (e < E) { s = src[e]; d = dst[e]; } else { s = d = e - E; }
    float w = ex[e * 4 + hd] / den[d * 4 + hd];
    atomicAdd(out + (size_t)d * C + ch, w * h[(size_t)s * C + ch]);
}

// --------------------------- LayerNorm (D=64) ------------------------------
// t = LN(t + a) * g + b ; writes fp32 master (in place) + fp16 mirror.
__global__ __launch_bounds__(128)
void add_ln(float* __restrict__ t, const float* __restrict__ a,
            const float* __restrict__ g, const float* __restrict__ b,
            _Float16* __restrict__ th, int n) {
    int row  = blockIdx.x * 4 + (threadIdx.x >> 5);   // n % 4 == 0 always
    int lane = threadIdx.x & 31;
    float* tp = t + (size_t)row * DMODEL;
    const float* ap = a + (size_t)row * DMODEL;
    _Float16* hp = th + (size_t)row * DMODEL;
    float v0 = tp[lane]      + ap[lane];
    float v1 = tp[lane + 32] + ap[lane + 32];
    float s  = v0 + v1;
    float s2 = v0 * v0 + v1 * v1;
#pragma unroll
    for (int o = 16; o > 0; o >>= 1) {
        s  += __shfl_xor(s,  o, 32);
        s2 += __shfl_xor(s2, o, 32);
    }
    float mean = s * (1.0f / 64.0f);
    float var  = s2 * (1.0f / 64.0f) - mean * mean;
    float inv  = rsqrtf(var + 1e-5f);
    float r0 = (v0 - mean) * inv * g[lane]      + b[lane];
    float r1 = (v1 - mean) * inv * g[lane + 32] + b[lane + 32];
    tp[lane]      = r0;          tp[lane + 32] = r1;
    hp[lane]      = (_Float16)r0; hp[lane + 32] = (_Float16)r1;
}

// ------------------------------ pooling / FC -------------------------------
__global__ void pool_sum(const float* __restrict__ t, const int* __restrict__ batch,
                         float* __restrict__ pooled, float* __restrict__ cnt, int n) {
    int i = blockIdx.x * blockDim.x + threadIdx.x;
    if (i >= n * DMODEL) return;
    int node = i >> 6, c = i & 63;
    int g = batch[node];
    atomicAdd(pooled + g * DMODEL + c, t[i]);
    if (c == 0) atomicAdd(cnt + g, 1.0f);
}

__global__ void final_fc(const float* __restrict__ pooled, const float* __restrict__ cnt,
                         const float* __restrict__ fcW, const float* __restrict__ fcb,
                         float* __restrict__ out) {
    int t = threadIdx.x;
    if (t >= NG * 2) return;
    int g = t >> 1, o = t & 1;
    float c = fmaxf(cnt[g], 1.0f);
    float s = 0.0f;
    for (int k = 0; k < DMODEL; ++k) s += (pooled[g * DMODEL + k] / c) * fcW[o * DMODEL + k];
    out[g * 2 + o] = s + fcb[o];
}

// ------------------------------- launcher ----------------------------------
extern "C" void kernel_launch(void* const* d_in, const int* in_sizes, int n_in,
                              void* d_out, int out_size, void* d_ws, size_t ws_size,
                              hipStream_t stream) {
    (void)in_sizes; (void)n_in; (void)out_size; (void)ws_size;
    const int N = N_NODES, E = N_EDGES, EE = N_EDGES + N_NODES;

    auto F = [&](int i) { return (const float*)d_in[i]; };
    const float* x    = F(0);
    const int*   ei   = (const int*)d_in[1];
    const int*   batch= (const int*)d_in[2];
    const float *c1W = F(3), *c1as = F(4), *c1ad = F(5), *c1b = F(6);
    const float *c2W = F(7), *c2as = F(8), *c2ad = F(9), *c2b = F(10);
    float* out = (float*)d_out;

    // ---- fp32 workspace carving ----
    float* buf0 = (float*)d_ws;                // N*256
    float* buf1 = buf0 + (size_t)N * 256;      // N*256
    float* ssrc = buf1 + (size_t)N * 256;      // N*4
    float* sdst = ssrc + N * 4;
    float* mbuf = sdst + N * 4;
    float* dbuf = mbuf + N * 4;
    float* exb  = dbuf + N * 4;                // EE*4
    float* pooled = exb + (size_t)EE * 4;      // NG*64
    float* cnt    = pooled + NG * DMODEL;      // NG

    // ---- fp16 workspace carving (after fp32 arena; total offset %16==0) ----
    _Float16* hptr = (_Float16*)(cnt + NG);
    auto halloc = [&](size_t nh) { _Float16* p = hptr; hptr += nh; return p; };
    _Float16* wt1h = halloc(256 * 128);
    _Float16* wt2h = halloc(64 * 256);
    _Float16* encW[2][4];                      // Wv, Wo, W1, W2
    for (int L = 0; L < 2; ++L) {
        encW[L][0] = halloc(64 * 64);
        encW[L][1] = halloc(64 * 64);
        encW[L][2] = halloc((size_t)FFDIM * 64);
        encW[L][3] = halloc((size_t)64 * FFDIM);
    }
    _Float16* decW[2][6];                      // Wv, Wo, W1, W2, Wv_x, Wo_x
    for (int L = 0; L < 2; ++L) {
        decW[L][0] = halloc(64 * 64);
        decW[L][1] = halloc(64 * 64);
        decW[L][2] = halloc((size_t)FFDIM * 64);
        decW[L][3] = halloc((size_t)64 * FFDIM);
        decW[L][4] = halloc(64 * 64);
        decW[L][5] = halloc(64 * 64);
    }
    _Float16* actA = halloc((size_t)N * 256);  // xh -> g1h -> ffn intermediate
    _Float16* th   = halloc((size_t)N * 64);
    _Float16* vh   = halloc((size_t)N * 64);
    _Float16* memh = halloc((size_t)N * 64);

    auto fill = [&](float* p, float v, long long n) {
        fill_f32<<<(unsigned)((n + 255) / 256), 256, 0, stream>>>(p, v, n);
    };
    auto cvt = [&](_Float16* d, const float* s, long long n) {
        cvt_f32_f16<<<(unsigned)((n + 255) / 256), 256, 0, stream>>>(d, s, n);
    };
    auto gemm_f = [&](const _Float16* A, const _Float16* B, float* C, const float* bias,
                      int M, int Nc, int K, int relu) {
        dim3 g(Nc / 64, M / 64);
        wmma_gemm_nt<float><<<g, 128, 0, stream>>>(A, B, C, bias, M, Nc, K, relu);
    };
    auto gemm_h = [&](const _Float16* A, const _Float16* B, _Float16* C, const float* bias,
                      int M, int Nc, int K, int relu) {
        dim3 g(Nc / 64, M / 64);
        wmma_gemm_nt<_Float16><<<g, 128, 0, stream>>>(A, B, C, bias, M, Nc, K, relu);
    };

    // ---- one-time weight conversions (per launch; deterministic) ----
    transpose_cvt<<<(128 * 256 + 255) / 256, 256, 0, stream>>>(wt1h, c1W, 128, 256);
    transpose_cvt<<<(256 * 64 + 255) / 256, 256, 0, stream>>>(wt2h, c2W, 256, 64);
    for (int L = 0; L < 2; ++L) {
        int bi = 11 + L * 12;
        cvt(encW[L][0], F(bi + 0) + 128 * 64, 64 * 64);         // Wv slice of Wqkv
        cvt(encW[L][1], F(bi + 2), 64 * 64);                    // Wo
        cvt(encW[L][2], F(bi + 4), (long long)FFDIM * 64);      // W1
        cvt(encW[L][3], F(bi + 6), (long long)64 * FFDIM);      // W2
    }
    for (int L = 0; L < 2; ++L) {
        int bi = 35 + L * 18;
        cvt(decW[L][0], F(bi + 0) + 128 * 64, 64 * 64);
        cvt(decW[L][1], F(bi + 2), 64 * 64);
        cvt(decW[L][2], F(bi + 4), (long long)FFDIM * 64);
        cvt(decW[L][3], F(bi + 6), (long long)64 * FFDIM);
        cvt(decW[L][4], F(bi + 12) + 128 * 64, 64 * 64);        // Wv_x slice
        cvt(decW[L][5], F(bi + 14), 64 * 64);                   // Wo_x
    }

    const float NEG_INF = -__builtin_huge_valf();

    // ---------------- GAT layer 1 (oc=64, N*256 features) ----------------
    cvt(actA, x, (long long)N * 128);                           // xh
    gemm_f(actA, wt1h, buf0, nullptr, N, 256, 128, 0);          // h = x@W
    gat_scores<<<(N * 4 + 255) / 256, 256, 0, stream>>>(buf0, c1as, c1ad, ssrc, sdst, N, 64);
    fill(mbuf, NEG_INF, N * 4);
    fill(dbuf, 0.0f, N * 4);
    fill(buf1, 0.0f, (long long)N * 256);
    edge_max<<<(EE * 4 + 255) / 256, 256, 0, stream>>>(ei, ei + E, ssrc, sdst, mbuf, E, N);
    edge_exp<<<(EE * 4 + 255) / 256, 256, 0, stream>>>(ei, ei + E, ssrc, sdst, mbuf, exb, dbuf, E, N);
    edge_aggr<<<(unsigned)(((long long)EE * 256 + 255) / 256), 256, 0, stream>>>(
        ei, ei + E, exb, dbuf, buf0, buf1, E, N, 64);
    bias_act<<<(unsigned)(((long long)N * 256 + 255) / 256), 256, 0, stream>>>(
        buf1, c1b, (long long)N * 256, 256, 0.01f);

    // ---------------- GAT layer 2 (oc=16, N*64 features) -----------------
    cvt(actA, buf1, (long long)N * 256);                        // g1h
    gemm_f(actA, wt2h, buf0, nullptr, N, 64, 256, 0);           // h2 = g1@W2
    float* g2 = buf0 + (size_t)N * 64;
    gat_scores<<<(N * 4 + 255) / 256, 256, 0, stream>>>(buf0, c2as, c2ad, ssrc, sdst, N, 16);
    fill(mbuf, NEG_INF, N * 4);
    fill(dbuf, 0.0f, N * 4);
    fill(g2, 0.0f, (long long)N * 64);
    edge_max<<<(EE * 4 + 255) / 256, 256, 0, stream>>>(ei, ei + E, ssrc, sdst, mbuf, E, N);
    edge_exp<<<(EE * 4 + 255) / 256, 256, 0, stream>>>(ei, ei + E, ssrc, sdst, mbuf, exb, dbuf, E, N);
    edge_aggr<<<(unsigned)(((long long)EE * 64 + 255) / 256), 256, 0, stream>>>(
        ei, ei + E, exb, dbuf, buf0, g2, E, N, 16);
    bias_act<<<(unsigned)(((long long)N * 64 + 255) / 256), 256, 0, stream>>>(
        g2, c2b, (long long)N * 64, 64, 0.01f);

    // ---------------- transformer (seq_len == 1) -------------------------
    float* tb = buf1;                          // fp32 master t
    float* ab = buf1 + (size_t)N * 64;         // fp32 attn/FFN output (residual addend)
    _Float16* ffnh = actA;                     // CHUNK*2048 <= N*256 halves
    hipMemcpyAsync(tb, g2, (size_t)N * 64 * sizeof(float), hipMemcpyDeviceToDevice, stream);
    cvt(th, g2, (long long)N * 64);

    auto run_ffn = [&](const _Float16* W1h, const float* b1,
                       const _Float16* W2h, const float* b2) {
        for (int off = 0; off < N; off += CHUNK) {
            int rows = (N - off < CHUNK) ? (N - off) : CHUNK;
            gemm_h(th + (size_t)off * 64, W1h, ffnh, b1, rows, FFDIM, 64, 1);
            gemm_f(ffnh, W2h, ab + (size_t)off * 64, b2, rows, 64, FFDIM, 0);
        }
    };

    // encoder layers (post-norm). MHA with S=T=1: a = (t@Wv^T+bv)@Wo^T+bo.
    for (int L = 0; L < 2; ++L) {
        int bi = 11 + L * 12;
        gemm_h(th, encW[L][0], vh, F(bi + 1) + 128, N, 64, 64, 0);            // v
        gemm_f(vh, encW[L][1], ab, F(bi + 3), N, 64, 64, 0);                  // o
        add_ln<<<N / 4, 128, 0, stream>>>(tb, ab, F(bi + 8), F(bi + 9), th, N);   // ln1
        run_ffn(encW[L][2], F(bi + 5), encW[L][3], F(bi + 7));
        add_ln<<<N / 4, 128, 0, stream>>>(tb, ab, F(bi + 10), F(bi + 11), th, N); // ln2
    }
    hipMemcpyAsync(memh, th, (size_t)N * 64 * sizeof(_Float16), hipMemcpyDeviceToDevice, stream);

    // decoder layers: self-attn, cross-attn (kv = mem), FFN
    for (int L = 0; L < 2; ++L) {
        int bi = 35 + L * 18;
        gemm_h(th, decW[L][0], vh, F(bi + 1) + 128, N, 64, 64, 0);
        gemm_f(vh, decW[L][1], ab, F(bi + 3), N, 64, 64, 0);
        add_ln<<<N / 4, 128, 0, stream>>>(tb, ab, F(bi + 8), F(bi + 9), th, N);   // ln1
        gemm_h(memh, decW[L][4], vh, F(bi + 13) + 128, N, 64, 64, 0);             // cross v
        gemm_f(vh, decW[L][5], ab, F(bi + 15), N, 64, 64, 0);                     // cross o
        add_ln<<<N / 4, 128, 0, stream>>>(tb, ab, F(bi + 10), F(bi + 11), th, N); // ln2
        run_ffn(decW[L][2], F(bi + 5), decW[L][3], F(bi + 7));
        add_ln<<<N / 4, 128, 0, stream>>>(tb, ab, F(bi + 16), F(bi + 17), th, N); // ln3
    }

    // ---------------- mean pool per graph + FC ---------------------------
    fill(pooled, 0.0f, NG * DMODEL);
    fill(cnt, 0.0f, NG);
    pool_sum<<<(N * DMODEL + 255) / 256, 256, 0, stream>>>(tb, batch, pooled, cnt, N);
    final_fc<<<1, 32, 0, stream>>>(pooled, cnt, F(71), F(72), out);
}
